// BitEditorLayer_35450660061817
// MI455X (gfx1250) — matrix operand order repaired and checked by reading, not compiled, not run on hardware
//
#include <hip/hip_runtime.h>

#define D_     1024
#define NH     16
#define HDIM   64
#define DFF_   2048
#define NE     8
#define NB_    2
#define NS_    2048
#define NT_    (NB_*NS_)          // 4096 tokens
#define CHUNK_ 128
#define NCH_   (NS_/CHUNK_)       // 16 chunks
#define NSCAN_ (2*NB_*NH)         // 64 independent scans (dir,b,h)
#define ROWCAP 9216               // 2T padded to 128 per expert segment
#define NMBLK  (ROWCAP/128)       // 72
#define LDT    40                 // LDS tile row stride (f16), 80B: conflict-free b128

typedef _Float16 f16;
typedef __attribute__((ext_vector_type(16))) _Float16 v16h;
typedef __attribute__((ext_vector_type(8)))  _Float16 v8h;
typedef __attribute__((ext_vector_type(8)))  float    v8f;
typedef __attribute__((ext_vector_type(4)))  int      v4i;

#define AS1 __attribute__((address_space(1)))
#define AS3 __attribute__((address_space(3)))

// Use CDNA5 async global->LDS copies (ASYNCcnt) when the toolchain exposes them.
#if defined(__has_builtin)
#if __has_builtin(__builtin_amdgcn_global_load_async_to_lds_b128) && \
    __has_builtin(__builtin_amdgcn_s_wait_asynccnt)
#define USE_ASYNC 1
#endif
#endif
#ifndef USE_ASYNC
#define USE_ASYNC 0
#endif

// copy 16B global -> LDS
__device__ __forceinline__ void cp16(const f16* g, f16* s) {
#if USE_ASYNC
  __builtin_amdgcn_global_load_async_to_lds_b128((AS1 v4i*)g, (AS3 v4i*)s, 0, 0);
#else
  *(uint4*)s = *(const uint4*)g;
#endif
}
__device__ __forceinline__ void wait_stage() {
#if USE_ASYNC
  __builtin_amdgcn_s_wait_asynccnt(0);
#endif
  __syncthreads();
}

// ---------------- WMMA fragment loaders (ISA 7.12.2 layouts) ----------------
// A 16x32 f16: lane m=L%16, half=L/16; element e<8 -> k=half*8+e ; e>=8 -> k=16+half*8+(e-8)
__device__ __forceinline__ v16h frag_a(const f16* tile, int row, int half) {
  const v8h lo = *(const v8h*)(tile + row * LDT + half * 8);
  const v8h hi = *(const v8h*)(tile + row * LDT + 16 + half * 8);
  return __builtin_shufflevector(lo, hi, 0,1,2,3,4,5,6,7,8,9,10,11,12,13,14,15);
}
// B 32x16 f16: lane n=L%16, half=L/16; element e<8 -> k=half*8+e ; e>=8 -> k=16+half*8+(e-8)
__device__ __forceinline__ v16h frag_b(const f16* tile, int row, int half) {
  const v8h lo = *(const v8h*)(tile + row * LDT + half * 8);
  const v8h hi = *(const v8h*)(tile + row * LDT + 16 + half * 8);
  return __builtin_shufflevector(lo, hi, 0,1,2,3,4,5,6,7,8,9,10,11,12,13,14,15);
}

// ---------------- unified double-buffered WMMA GEMM ----------------------------
// MODE 0: Cf[M,N] = A[M,K]*Bt[N,K]^T (+addend if non-null, hoisted)
// MODE 1: (MoE up)  A rows gathered via rowmap; Ch = f16(silu(acc)); Bt per expert
// MODE 2: (MoE down) dense A; Cf = acc; Bt per expert
template <int MODE>
__global__ void __launch_bounds__(256) gemm_wmma(
    const f16* __restrict__ A, int lda,
    const f16* __restrict__ Bt, int ldb,
    float* __restrict__ Cf, f16* __restrict__ Ch, int ldc,
    const float* __restrict__ addend, int K,
    const int* __restrict__ rowmap, const int* __restrict__ blkExpert) {
  if (MODE > 0) { if (blkExpert[blockIdx.y] < 0) return; }
  __shared__ f16 As[2][128 * LDT];
  __shared__ f16 Bs[2][128 * LDT];
  const int tid = threadIdx.x, lane = tid & 31;
  const int wave = tid >> 5, wm = wave >> 2, wn = wave & 3;
  const int mBase = blockIdx.y * 128, nBase = blockIdx.x * 128;
  const int laneM = lane & 15, half = lane >> 4;

  const f16* Bte = Bt;
  if (MODE > 0) Bte += (size_t)blkExpert[blockIdx.y] * ((size_t)DFF_ * D_);

  // staging addresses: 2 threads per tile row, 32B (2 x b128) each
  const int srow = tid >> 1, sseg = tid & 1;
  int arow = mBase + srow;
  if (MODE == 1) {               // gather; padded rows read token 0 (row-local, never consumed)
    int tk = rowmap[arow];
    arow = tk < 0 ? 0 : tk;
  }
  const f16* gA = A + (size_t)arow * lda + sseg * 16;
  const f16* gB = Bte + (size_t)(nBase + srow) * ldb + sseg * 16;
  f16* sA0 = &As[0][srow * LDT + sseg * 16];
  f16* sA1 = &As[1][srow * LDT + sseg * 16];
  f16* sB0 = &Bs[0][srow * LDT + sseg * 16];
  f16* sB1 = &Bs[1][srow * LDT + sseg * 16];

  v8f acc[4][2];
  for (int i = 0; i < 4; i++) for (int j = 0; j < 2; j++)
    for (int r = 0; r < 8; r++) acc[i][j][r] = 0.f;

  // prologue: stage K-step 0 into buffer 0
  cp16(gA, sA0); cp16(gA + 8, sA0 + 8);
  cp16(gB, sB0); cp16(gB + 8, sB0 + 8);
  wait_stage();

  int buf = 0;
  for (int k0 = 0; k0 < K; k0 += 32) {
    // stage next K-step into the other buffer while computing on this one
    if (k0 + 32 < K) {
      f16* dA = buf ? sA0 : sA1;
      f16* dB = buf ? sB0 : sB1;
      cp16(gA + k0 + 32, dA); cp16(gA + k0 + 40, dA + 8);
      cp16(gB + k0 + 32, dB); cp16(gB + k0 + 40, dB + 8);
    }
    const f16* At = As[buf];
    const f16* Bti = Bs[buf];
    v16h bf0 = frag_b(Bti, wn * 32 + laneM, half);
    v16h bf1 = frag_b(Bti, wn * 32 + 16 + laneM, half);
#pragma unroll
    for (int fm = 0; fm < 4; fm++) {
      v16h af = frag_a(At, wm * 64 + fm * 16 + laneM, half);
      acc[fm][0] = __builtin_amdgcn_wmma_f32_16x16x32_f16(
          false, af, false, bf0, (short)0, acc[fm][0], false, false);
      acc[fm][1] = __builtin_amdgcn_wmma_f32_16x16x32_f16(
          false, af, false, bf1, (short)0, acc[fm][1], false, false);
    }
    wait_stage();
    buf ^= 1;
  }

  // epilogue (uniform branches hoisted out of the store loops)
  if (MODE == 0) {
    if (addend != nullptr) {
      for (int fm = 0; fm < 4; fm++) for (int fn = 0; fn < 2; fn++) {
        int row0 = mBase + wm * 64 + fm * 16 + half * 8;
        int col  = nBase + wn * 32 + fn * 16 + laneM;
#pragma unroll
        for (int r = 0; r < 8; r++) {
          size_t idx = (size_t)(row0 + r) * ldc + col;
          Cf[idx] = acc[fm][fn][r] + addend[idx];
        }
      }
    } else {
      for (int fm = 0; fm < 4; fm++) for (int fn = 0; fn < 2; fn++) {
        int row0 = mBase + wm * 64 + fm * 16 + half * 8;
        int col  = nBase + wn * 32 + fn * 16 + laneM;
#pragma unroll
        for (int r = 0; r < 8; r++)
          Cf[(size_t)(row0 + r) * ldc + col] = acc[fm][fn][r];
      }
    }
  } else if (MODE == 1) {
    for (int fm = 0; fm < 4; fm++) for (int fn = 0; fn < 2; fn++) {
      int row0 = mBase + wm * 64 + fm * 16 + half * 8;
      int col  = nBase + wn * 32 + fn * 16 + laneM;
#pragma unroll
      for (int r = 0; r < 8; r++) {
        float v = acc[fm][fn][r];
        Ch[(size_t)(row0 + r) * ldc + col] = (f16)(v / (1.f + __expf(-v)));  // silu
      }
    }
  } else {
    for (int fm = 0; fm < 4; fm++) for (int fn = 0; fn < 2; fn++) {
      int row0 = mBase + wm * 64 + fm * 16 + half * 8;
      int col  = nBase + wn * 32 + fn * 16 + laneM;
#pragma unroll
      for (int r = 0; r < 8; r++)
        Cf[(size_t)(row0 + r) * ldc + col] = acc[fm][fn][r];
    }
  }
}

// ---------------- small kernels -------------------------------------------------
__global__ void zero_k(int* p, int n) {
  int i = blockIdx.x * 256 + threadIdx.x;
  if (i < n) p[i] = 0;
}

__global__ void rmsnorm_k(const float* __restrict__ x, const float* __restrict__ w,
                          f16* __restrict__ out) {
  const int t = blockIdx.x;
  __shared__ float red[256];
  const float* xr = x + (size_t)t * D_;
  float s = 0.f;
  for (int i = threadIdx.x; i < D_; i += 256) { float v = xr[i]; s += v * v; }
  red[threadIdx.x] = s; __syncthreads();
  for (int o = 128; o > 0; o >>= 1) {
    if (threadIdx.x < o) red[threadIdx.x] += red[threadIdx.x + o];
    __syncthreads();
  }
  const float rs = rsqrtf(red[0] / (float)D_ + 1e-6f);
  for (int i = threadIdx.x; i < D_; i += 256)
    out[(size_t)t * D_ + i] = (f16)(xr[i] * rs * w[i]);
}

__global__ void conv_qkv(const float* __restrict__ Wr, const float* __restrict__ Wk,
                         const float* __restrict__ Wv, f16* __restrict__ wt) {
  size_t i = (size_t)blockIdx.x * 256 + threadIdx.x;     // 3072*1024 elements
  int n = (int)(i >> 10), k = (int)(i & 1023);
  const float* W = (n < 1024) ? Wr : ((n < 2048) ? Wk : Wv);
  wt[i] = (f16)W[(size_t)k * 1024 + (n & 1023)];
}

__global__ void conv_t(const float* __restrict__ W, f16* __restrict__ wt, int N, int K) {
  size_t i = (size_t)blockIdx.x * 256 + threadIdx.x;
  if (i >= (size_t)N * K) return;
  int n = (int)(i / K), k = (int)(i % K);
  wt[i] = (f16)W[(size_t)k * N + n];
}

// ---------------- chunked bidirectional RWKV scan ------------------------------
// Pass A: per-chunk local decayed kv sum   L[k][v] = sum_j w_k^{C-1-j} k_j[k] v_j[v]
__global__ void __launch_bounds__(64) scan_chunk_local(const float* __restrict__ rkv,
    const unsigned char* __restrict__ mask, const float* __restrict__ decay,
    float* __restrict__ chunkL) {
  const int sc = blockIdx.y, c = blockIdx.x, tid = threadIdx.x;   // tid = v col
  const int dir = sc >> 5, b = (sc >> 4) & 1, h = sc & 15;
  __shared__ float ksh[64], wsh[64];
  wsh[tid] = __expf(-__expf(decay[h * 64 + tid]));
  float L[64];
#pragma unroll
  for (int k = 0; k < 64; k++) L[k] = 0.f;
  for (int j = 0; j < CHUNK_; j++) {
    int s = dir ? (NS_ - 1 - (c * CHUNK_ + j)) : (c * CHUNK_ + j);
    int t = b * NS_ + s;
    float m = mask[t] ? 1.f : 0.f;
    __syncthreads();
    ksh[tid] = rkv[(size_t)t * 3072 + 1024 + h * 64 + tid] * m;
    __syncthreads();
    float vv = rkv[(size_t)t * 3072 + 2048 + h * 64 + tid] * m;
#pragma unroll
    for (int k = 0; k < 64; k++) L[k] = wsh[k] * L[k] + ksh[k] * vv;
  }
  float* out = chunkL + ((size_t)sc * NCH_ + c) * 4096;
#pragma unroll
  for (int k = 0; k < 64; k++) out[k * 64 + tid] = L[k];
}

// Pass B: inter-chunk recurrence S_{c+1} = w^C * S_c + L_c ; store chunk-start states
__global__ void __launch_bounds__(256) scan_chunk_states(const float* __restrict__ decay,
    const float* __restrict__ chunkL, float* __restrict__ chunkS) {
  const int sc = blockIdx.x, tid = threadIdx.x;
  const int h = sc & 15;
  float Sreg[16], wc[16];
#pragma unroll
  for (int i = 0; i < 16; i++) {
    int e = i * 256 + tid, k = e >> 6;
    wc[i] = __expf(-(float)CHUNK_ * __expf(decay[h * 64 + k]));
    Sreg[i] = 0.f;
  }
  for (int c = 0; c < NCH_; c++) {
    const float* Lc = chunkL + ((size_t)sc * NCH_ + c) * 4096;
    float* Sc = chunkS + ((size_t)sc * NCH_ + c) * 4096;
#pragma unroll
    for (int i = 0; i < 16; i++) {
      int e = i * 256 + tid;
      Sc[e] = Sreg[i];
      Sreg[i] = wc[i] * Sreg[i] + Lc[e];
    }
  }
}

// Pass C: per-chunk outputs starting from chunk-start state
__global__ void __launch_bounds__(64) scan_chunk_out(const float* __restrict__ rkv,
    const unsigned char* __restrict__ mask, const float* __restrict__ decay,
    const float* __restrict__ bonus, const float* __restrict__ chunkS,
    float* __restrict__ ybuf) {
  const int sc = blockIdx.y, c = blockIdx.x, tid = threadIdx.x;   // tid = v col
  const int dir = sc >> 5, b = (sc >> 4) & 1, h = sc & 15;
  __shared__ float rsh[64], ksh[64], wsh[64], ush[64];
  wsh[tid] = __expf(-__expf(decay[h * 64 + tid]));
  ush[tid] = bonus[h * 64 + tid];
  float Scol[64];
  const float* S0 = chunkS + ((size_t)sc * NCH_ + c) * 4096;
#pragma unroll
  for (int k = 0; k < 64; k++) Scol[k] = S0[k * 64 + tid];
  for (int j = 0; j < CHUNK_; j++) {
    int s = dir ? (NS_ - 1 - (c * CHUNK_ + j)) : (c * CHUNK_ + j);
    int t = b * NS_ + s;
    float m = mask[t] ? 1.f : 0.f;
    __syncthreads();
    rsh[tid] = rkv[(size_t)t * 3072 + h * 64 + tid];
    ksh[tid] = rkv[(size_t)t * 3072 + 1024 + h * 64 + tid] * m;
    __syncthreads();
    float vv = rkv[(size_t)t * 3072 + 2048 + h * 64 + tid] * m;
    float y = 0.f;
#pragma unroll
    for (int k = 0; k < 64; k++) {
      float kv = ksh[k] * vv;
      y += rsh[k] * (Scol[k] + ush[k] * kv);
      Scol[k] = wsh[k] * Scol[k] + kv;
    }
    ybuf[((size_t)dir * NT_ + t) * 1024 + h * 64 + tid] = y;
  }
}

__global__ void combine_y(const float* __restrict__ ybuf, f16* __restrict__ yf16) {
  size_t i = (size_t)blockIdx.x * 256 + threadIdx.x;
  yf16[i] = (f16)(ybuf[i] + ybuf[(size_t)NT_ * 1024 + i]);
}

// ---------------- router / MoE bookkeeping -------------------------------------
__global__ void router_k(const f16* __restrict__ xn2, const float* __restrict__ rw,
                         float* __restrict__ probs, float* __restrict__ gates,
                         int* __restrict__ topi, int* __restrict__ counts) {
  const int t = blockIdx.x, tid = threadIdx.x;    // 128 threads
  __shared__ float red[128][8];
  float p[8];
#pragma unroll
  for (int e = 0; e < 8; e++) p[e] = 0.f;
  for (int d = tid; d < D_; d += 128) {
    float xv = (float)xn2[(size_t)t * D_ + d];
#pragma unroll
    for (int e = 0; e < 8; e++) p[e] += xv * rw[d * 8 + e];
  }
#pragma unroll
  for (int e = 0; e < 8; e++) red[tid][e] = p[e];
  __syncthreads();
  for (int o = 64; o > 0; o >>= 1) {
    if (tid < o)
      for (int e = 0; e < 8; e++) red[tid][e] += red[tid + o][e];
    __syncthreads();
  }
  if (tid == 0) {
    float pr[8], mx = red[0][0];
    for (int e = 1; e < 8; e++) mx = fmaxf(mx, red[0][e]);
    float sum = 0.f;
    for (int e = 0; e < 8; e++) { pr[e] = __expf(red[0][e] - mx); sum += pr[e]; }
    for (int e = 0; e < 8; e++) { pr[e] /= sum; probs[t * 8 + e] = pr[e]; }
    int i0 = 0;
    for (int e = 1; e < 8; e++) if (pr[e] > pr[i0]) i0 = e;
    int i1 = (i0 == 0) ? 1 : 0;
    for (int e = 0; e < 8; e++) if (e != i0 && pr[e] > pr[i1]) i1 = e;
    float gs = pr[i0] + pr[i1];
    gates[t * 2] = pr[i0] / gs; gates[t * 2 + 1] = pr[i1] / gs;
    topi[t * 2] = i0; topi[t * 2 + 1] = i1;
    atomicAdd(&counts[i0], 1); atomicAdd(&counts[i1], 1);
  }
}

__global__ void psum_k(const float* __restrict__ probs, float* __restrict__ psum) {
  const int e = blockIdx.x, tid = threadIdx.x;
  __shared__ float red[256];
  float s = 0.f;
  for (int t = tid; t < NT_; t += 256) s += probs[t * 8 + e];
  red[tid] = s; __syncthreads();
  for (int o = 128; o > 0; o >>= 1) {
    if (tid < o) red[tid] += red[tid + o];
    __syncthreads();
  }
  if (tid == 0) psum[e] = red[0];
}

__global__ void build_meta(const int* __restrict__ counts, int* __restrict__ off,
                           int* __restrict__ blkExpert) {
  if (threadIdx.x == 0) {
    int o = 0;
    for (int e = 0; e < 8; e++) { off[e] = o; o += ((counts[e] + 127) / 128) * 128; }
    off[8] = o;
    for (int i = 0; i < NMBLK; i++) {
      int rb = i * 128, ex = -1;
      for (int e = 0; e < 8; e++) if (rb >= off[e] && rb < off[e + 1]) ex = e;
      blkExpert[i] = ex;
    }
  }
}

__global__ void init_rowmap(int* __restrict__ rowmap) {
  int i = blockIdx.x * 256 + threadIdx.x;
  if (i < ROWCAP) rowmap[i] = -1;
}

__global__ void scatter_k(const int* __restrict__ topi, const int* __restrict__ off,
                          int* __restrict__ fill, int* __restrict__ rowmap,
                          int* __restrict__ pos_of) {
  int t = blockIdx.x * 256 + threadIdx.x;
  if (t >= NT_) return;
  for (int s = 0; s < 2; s++) {
    int e = topi[t * 2 + s];
    int p = atomicAdd(&fill[e], 1);
    int row = off[e] + p;
    rowmap[row] = t;
    pos_of[t * 2 + s] = row;
  }
}

__global__ void final_k(const float* __restrict__ x1, const float* __restrict__ outpair,
                        const float* __restrict__ gates, const int* __restrict__ pos_of,
                        float* __restrict__ out) {
  const int t = blockIdx.y;
  const int d = blockIdx.x * 256 + threadIdx.x;
  size_t idx = (size_t)t * D_ + d;
  int r0 = pos_of[t * 2], r1 = pos_of[t * 2 + 1];
  out[idx] = x1[idx] + gates[t * 2] * outpair[(size_t)r0 * D_ + d]
                     + gates[t * 2 + 1] * outpair[(size_t)r1 * D_ + d];
}

__global__ void aux_k(const int* __restrict__ counts, const float* __restrict__ psum,
                      float* __restrict__ out) {
  if (threadIdx.x == 0) {
    float a = 0.f;
    for (int e = 0; e < 8; e++) {
      float frac = (float)counts[e] / (float)(NT_ * 2);
      float pm = psum[e] / (float)NT_;
      a += frac * pm;
    }
    out[0] = 8.f * a;
  }
}

// ---------------- workspace layout (bytes) -------------------------------------
#define OFF_WQKV   ((size_t)0)                       // 3072*1024 f16
#define OFF_WOT    (OFF_WQKV + 6291456)              // 1024*1024 f16
#define OFF_W1T    (OFF_WOT + 2097152)               // 8*2048*1024 f16
#define OFF_W2T    (OFF_W1T + 33554432)              // 8*1024*2048 f16
#define OFF_XN1    (OFF_W2T + 33554432)              // 4096*1024 f16
#define OFF_YF16   (OFF_XN1 + 8388608)
#define OFF_XN2    (OFF_YF16 + 8388608)
#define OFF_X1     (OFF_XN2 + 8388608)               // 4096*1024 f32
#define OFF_YBUF   (OFF_X1 + 16777216)               // 2*4096*1024 f32
#define OFF_PROBS  (OFF_YBUF + 33554432)             // 4096*8 f32
#define OFF_GATES  (OFF_PROBS + 131072)              // 4096*2 f32
#define OFF_POS    (OFF_GATES + 32768)               // 4096*2 i32
#define OFF_TOPI   (OFF_POS + 32768)                 // 4096*2 i32
#define OFF_SMALL  (OFF_TOPI + 32768)                // counters etc (4KB)
#define OFF_ROWMAP (OFF_SMALL + 4096)                // 9216 i32
#define OFF_REGION (OFF_ROWMAP + 36864)              // union region (84MB)
// region sub-layout 1 (attention phase): rkv(50.3MB) | chunkL(16.8MB) | chunkS(16.8MB)
// region sub-layout 2 (MoE phase):       H(37.7MB)   | out_pair(37.7MB)

extern "C" void kernel_launch(void* const* d_in, const int* in_sizes, int n_in,
                              void* d_out, int out_size, void* d_ws, size_t ws_size,
                              hipStream_t stream) {
  (void)in_sizes; (void)n_in; (void)out_size; (void)ws_size;
  const float* x    = (const float*)d_in[0];
  const unsigned char* mask = (const unsigned char*)d_in[1];   // jax bool = 1 byte
  const float* n1w  = (const float*)d_in[2];
  const float* n2w  = (const float*)d_in[3];
  const float* Wr   = (const float*)d_in[4];
  const float* Wk   = (const float*)d_in[5];
  const float* Wv   = (const float*)d_in[6];
  const float* Wo   = (const float*)d_in[7];
  const float* dec  = (const float*)d_in[8];
  const float* bon  = (const float*)d_in[9];
  const float* rw   = (const float*)d_in[10];
  const float* w1   = (const float*)d_in[11];
  const float* w2   = (const float*)d_in[12];

  char* ws = (char*)d_ws;
  f16*   wqkv   = (f16*)(ws + OFF_WQKV);
  f16*   wot    = (f16*)(ws + OFF_WOT);
  f16*   w1t    = (f16*)(ws + OFF_W1T);
  f16*   w2t    = (f16*)(ws + OFF_W2T);
  f16*   xn1    = (f16*)(ws + OFF_XN1);
  f16*   yf16   = (f16*)(ws + OFF_YF16);
  f16*   xn2    = (f16*)(ws + OFF_XN2);
  float* x1     = (float*)(ws + OFF_X1);
  float* ybuf   = (float*)(ws + OFF_YBUF);
  float* probs  = (float*)(ws + OFF_PROBS);
  float* gates  = (float*)(ws + OFF_GATES);
  int*   pos_of = (int*)(ws + OFF_POS);
  int*   topi   = (int*)(ws + OFF_TOPI);
  int*   small  = (int*)(ws + OFF_SMALL);
  int*   counts = small + 0;
  int*   fill   = small + 8;
  int*   off    = small + 16;
  int*   blkexp = small + 32;
  float* psum   = (float*)(small + 112);
  int*   rowmap = (int*)(ws + OFF_ROWMAP);
  float* rkv    = (float*)(ws + OFF_REGION);
  float* chunkL = (float*)(ws + OFF_REGION + 50331648);
  float* chunkS = (float*)(ws + OFF_REGION + 67108864);
  f16*   Hbuf   = (f16*)(ws + OFF_REGION);
  float* outpr  = (float*)(ws + OFF_REGION + 37748736);

  // 0) zero small counters (counts/fill/off/blkexp/psum)
  zero_k<<<4, 256, 0, stream>>>(small, 1024);
  // 1) weight conversion / transpose to f16 K-contiguous layouts
  conv_qkv<<<12288, 256, 0, stream>>>(Wr, Wk, Wv, wqkv);
  conv_t<<<4096, 256, 0, stream>>>(Wo, wot, 1024, 1024);
  for (int e = 0; e < 8; e++) {
    conv_t<<<8192, 256, 0, stream>>>(w1 + (size_t)e * D_ * DFF_,
                                     w1t + (size_t)e * DFF_ * D_, DFF_, D_);
    conv_t<<<8192, 256, 0, stream>>>(w2 + (size_t)e * DFF_ * D_,
                                     w2t + (size_t)e * D_ * DFF_, D_, DFF_);
  }
  // 2) pre-norm + fused QKV WMMA GEMM: rkv[t][0:1024]=r, [1024:2048]=k, [2048:3072]=v
  rmsnorm_k<<<NT_, 256, 0, stream>>>(x, n1w, xn1);
  gemm_wmma<0><<<dim3(24, 32), 256, 0, stream>>>(xn1, 1024, wqkv, 1024, rkv, nullptr,
                                                 3072, nullptr, 1024, nullptr, nullptr);
  // 3) chunked bidirectional RWKV scan
  scan_chunk_local<<<dim3(NCH_, NSCAN_), 64, 0, stream>>>(rkv, mask, dec, chunkL);
  scan_chunk_states<<<NSCAN_, 256, 0, stream>>>(dec, chunkL, chunkS);
  scan_chunk_out<<<dim3(NCH_, NSCAN_), 64, 0, stream>>>(rkv, mask, dec, bon, chunkS, ybuf);
  combine_y<<<16384, 256, 0, stream>>>(ybuf, yf16);
  // 4) Wo WMMA GEMM fused with residual add: x1 = x + y @ Wo
  gemm_wmma<0><<<dim3(8, 32), 256, 0, stream>>>(yf16, 1024, wot, 1024, x1, nullptr,
                                                1024, x, 1024, nullptr, nullptr);
  // 5) second norm, router, aux bookkeeping
  rmsnorm_k<<<NT_, 256, 0, stream>>>(x1, n2w, xn2);
  router_k<<<NT_, 128, 0, stream>>>(xn2, rw, probs, gates, topi, counts);
  psum_k<<<8, 256, 0, stream>>>(probs, psum);
  build_meta<<<1, 32, 0, stream>>>(counts, off, blkexp);
  init_rowmap<<<(ROWCAP + 255) / 256, 256, 0, stream>>>(rowmap);
  scatter_k<<<(NT_ + 255) / 256, 256, 0, stream>>>(topi, off, fill, rowmap, pos_of);
  // 6) top-2 routed MoE via grouped WMMA GEMMs
  gemm_wmma<1><<<dim3(16, NMBLK), 256, 0, stream>>>(xn2, 1024, w1t, 1024, nullptr, Hbuf,
                                                    2048, nullptr, 1024, rowmap, blkexp);
  gemm_wmma<2><<<dim3(8, NMBLK), 256, 0, stream>>>(Hbuf, 2048, w2t, 2048, outpr, nullptr,
                                                   1024, nullptr, 2048, nullptr, blkexp);
  // 7) gated combine + residual, aux loss
  final_k<<<dim3(4, NT_), 256, 0, stream>>>(x1, outpr, gates, pos_of, (float*)d_out);
  aux_k<<<1, 32, 0, stream>>>(counts, psum, (float*)d_out + (size_t)NT_ * D_);
}
